// TextLSTMBinaryClassifier_74062416052929
// MI455X (gfx1250) — compile-verified, matrix operand order checked by reading, not run on hardware
//
#include <hip/hip_runtime.h>

// ---------------------------------------------------------------------------
// TextLSTMBinaryClassifier on gfx1250 (MI455X), wave32 + WMMA bf16.
//
// Sizes (fixed by reference): B=128, S=512, E=256, H=256, 4H=1024, V=800.
//
// Per (layer,direction) step:  g[16,1024] = [x_t, h] (16 x KTOT) @ Wcat^T + b
// computed with v_wmma_f32_16x16x32_bf16; gates evaluated in registers.
// Weights stream from L2 each step (asm clobber defeats LICM/spill); all hot
// global accesses go through address_space(1) so they lower to global_load/
// global_store (not FLAT), keeping LOADcnt separate from DScnt.  K-chunks are
// ping-pong double-buffered so weight-load latency hides under WMMA.
// ---------------------------------------------------------------------------

typedef __attribute__((ext_vector_type(16))) __bf16        v16bf;
typedef __attribute__((ext_vector_type(8)))  float         v8f;
typedef __attribute__((ext_vector_type(4)))  unsigned int  v4u;   // trivially
// copyable 16B vector (HIP's uint4 is a class; its copy-ctor cannot bind an
// address_space(1) reference, so it cannot be loaded through an AS1 pointer).

union Frag { v4u u[2]; v16bf v; };

#define B_   128
#define S_   512
#define E_   256
#define H_   256
#define FOURH 1024

// ---- global-address-space access helpers (force global_load/global_store) --
__device__ __forceinline__ v4u gload_u4(const void* p) {
  return *(const __attribute__((address_space(1))) v4u*)(unsigned long long)p;
}
__device__ __forceinline__ void gstore_b16(void* p, unsigned short v) {
  *(__attribute__((address_space(1))) unsigned short*)(unsigned long long)p = v;
}
__device__ __forceinline__ void gstore_f32(void* p, float v) {
  *(__attribute__((address_space(1))) float*)(unsigned long long)p = v;
}

__device__ __forceinline__ unsigned short f2bf(float f) {
  unsigned int u = __builtin_bit_cast(unsigned int, f);
  unsigned int r = (u + 0x7FFFu + ((u >> 16) & 1u)) >> 16;
  return (unsigned short)r;
}

__device__ __forceinline__ float sigm(float x)  { return 1.0f / (1.0f + __expf(-x)); }
__device__ __forceinline__ float tanh_f(float x){ return 1.0f - 2.0f / (__expf(2.0f * x) + 1.0f); }

// --------------------------- embedding lookup ------------------------------
__global__ void embed_kernel(const int* __restrict__ x,
                             const float* __restrict__ emb,
                             unsigned short* __restrict__ xe) {
  long long idx = (long long)blockIdx.x * blockDim.x + threadIdx.x;
  const long long total = (long long)B_ * S_ * (E_ / 4);
  if (idx >= total) return;
  long long tok_i = idx / (E_ / 4);
  int e4 = (int)(idx % (E_ / 4)) * 4;
  int tok = x[tok_i];
  const float4 f = *(const float4*)&emb[(long long)tok * E_ + e4];
  ushort4 o;
  o.x = f2bf(f.x); o.y = f2bf(f.y); o.z = f2bf(f.z); o.w = f2bf(f.w);
  *(ushort4*)&xe[idx * 4] = o;
}

// --------------------- weight / bias packing to bf16 -----------------------
// dst[n][k], n in [0,1024), k in [0,KX+256): k<KX -> wih[n][k], else whh[n][k-KX]
__global__ void pack_w_kernel(const float* __restrict__ wih,
                              const float* __restrict__ whh,
                              unsigned short* __restrict__ dst, int KX) {
  const int Ktot = KX + 256;
  const long long total = (long long)FOURH * Ktot;
  for (long long idx = (long long)blockIdx.x * blockDim.x + threadIdx.x;
       idx < total; idx += (long long)gridDim.x * blockDim.x) {
    int n = (int)(idx / Ktot);
    int k = (int)(idx % Ktot);
    float v = (k < KX) ? wih[(long long)n * KX + k]
                       : whh[(long long)n * 256 + (k - KX)];
    dst[idx] = f2bf(v);
  }
}

__global__ void pack_b_kernel(const float* __restrict__ bih,
                              const float* __restrict__ bhh,
                              float* __restrict__ dst) {
  int i = blockIdx.x * blockDim.x + threadIdx.x;
  if (i < FOURH) dst[i] = bih[i] + bhh[i];
}

// ----------------------------- LSTM direction ------------------------------
// grid: (8 batch tiles, 2 directions); block: 512 threads = 16 waves.
// Wave w owns gate-column tiles {w, 16+w, 32+w, 48+w} -> i/f/g/o for the same
// 16 h-columns; gate math stays in registers.  KX = input width (256 or 512).
template <int KX, bool WRITE_Y>
__global__ __launch_bounds__(512, 1)
void lstm_dir_kernel(const unsigned short* __restrict__ xin,   // [B][S][KX] bf16
                     const unsigned short* __restrict__ W0,    // fwd  [1024][KX+256] bf16
                     const unsigned short* __restrict__ W1,    // bwd
                     const float* __restrict__ bias0,          // fwd  [1024]
                     const float* __restrict__ bias1,          // bwd
                     unsigned short* __restrict__ y,           // [B][S][512] bf16 (WRITE_Y)
                     float* __restrict__ hfin) {               // [B][512] f32  (!WRITE_Y)
  constexpr int KTOT   = KX + 256;
  constexpr int NCH    = KTOT / 32;        // K-chunks per step (16 or 24)
  constexpr int STRIDE = KTOT + 16;        // +32B row pad (bank skew, keeps 16B align)
  __shared__ unsigned short abuf[16][STRIDE];   // [x_t | h] per batch-tile row

  const int dir   = blockIdx.y;
  const int rBase = blockIdx.x * 16;
  const unsigned short* __restrict__ W    = dir ? W1 : W0;
  const float* __restrict__          bias = dir ? bias1 : bias0;

  const int tid  = threadIdx.x;
  const int wv   = tid >> 5;        // wave 0..15
  const int lane = tid & 31;
  const int lr   = lane & 15;       // A row / B col / gate col within tile
  const int hi   = lane >> 4;       // K-half select
  const int c    = wv * 16 + lr;    // h column 0..255 owned by this lane

  const float bi = bias[c], bf_ = bias[256 + c], bg = bias[512 + c], bo = bias[768 + c];

  float cst[8];
#pragma unroll
  for (int i = 0; i < 8; ++i) cst[i] = 0.0f;

  // h_0 = 0 (ordered before first WMMA by the post-staging barrier below)
  {
    const v4u vz = {0u, 0u, 0u, 0u};
    for (int ch = tid; ch < 16 * 256 / 8; ch += 512) {
      int m = ch / 32, off = (ch & 31) * 8;
      *(v4u*)&abuf[m][KX + off] = vz;
    }
  }

  // per-lane element offsets of the 4 owned weight rows (i/f/g/o tiles)
  unsigned int rowoff[4];
#pragma unroll
  for (int g = 0; g < 4; ++g)
    rowoff[g] = (unsigned int)(g * 256 + wv * 16 + lr) * (unsigned int)KTOT;

  const v8f vzero = {0.f, 0.f, 0.f, 0.f, 0.f, 0.f, 0.f, 0.f};

  for (int it = 0; it < S_; ++it) {
    const int t = dir ? (S_ - 1 - it) : it;

    // stage x_t rows into abuf[*][0..KX)
    {
      constexpr int rowch = KX / 8;
      for (int ch = tid; ch < 16 * rowch; ch += 512) {
        int m = ch / rowch, off = (ch % rowch) * 8;
        const v4u v = gload_u4(&xin[((size_t)(rBase + m) * S_ + t) * KX + off]);
        *(v4u*)&abuf[m][off] = v;
      }
      if (it + 1 < S_) {  // hint next step's activations toward the caches
        const int tn = dir ? (S_ - 2 - it) : (it + 1);
        __builtin_prefetch(&xin[((size_t)(rBase + (tid & 15)) * S_ + tn) * KX], 0, 1);
      }
    }
    __syncthreads();     // x staged; h from previous step visible

    // Opaque copy of the weight base pointer: defeats LICM across the time
    // loop (otherwise LLVM pre-loads all weight fragments and spills them to
    // scratch).  Weights are re-fetched from L2 every step, as intended.
    unsigned long long wopq = (unsigned long long)W;
    asm volatile("" : "+s"(wopq));
    const unsigned short* Wt = (const unsigned short*)wopq;

    // ---- WMMA phase: acc[g] = [x_t,h](16xKTOT) @ Wcat^T tile (KTOTx16) ----
    v8f acc[4];
#pragma unroll
    for (int g = 0; g < 4; ++g) acc[g] = vzero;

    Frag A[2];        // ping-pong A fragments   (LDS)
    Frag Bf[2][4];    // ping-pong B fragments   (global/L2)

    auto load_chunk = [&](int buf, int kb) {
      // A fragment: 16-bit layout: lane=row; halves K { kb+8*hi+0..7, kb+16+8*hi+0..7 }
      A[buf].u[0] = *(const v4u*)&abuf[lr][kb + 8 * hi];
      A[buf].u[1] = *(const v4u*)&abuf[lr][kb + 16 + 8 * hi];
#pragma unroll
      for (int g = 0; g < 4; ++g) {
        const unsigned short* wr = Wt + rowoff[g] + kb;
        Bf[buf][g].u[0] = gload_u4(&wr[8 * hi]);
        Bf[buf][g].u[1] = gload_u4(&wr[16 + 8 * hi]);
      }
    };

    load_chunk(0, 0);
#pragma unroll
    for (int i = 0; i < NCH; ++i) {
      const int cur = i & 1;
      if (i + 1 < NCH) load_chunk(cur ^ 1, (i + 1) * 32);   // prefetch next chunk
#pragma unroll
      for (int g = 0; g < 4; ++g)
        acc[g] = __builtin_amdgcn_wmma_f32_16x16x32_bf16(
            false, A[cur].v, false, Bf[cur][g].v, (short)0, acc[g], false, false);
    }
    __syncthreads();     // all abuf reads done before h overwrite

    // ---- gates in registers: C element (m = i+8*hi, col c) ----------------
#pragma unroll
    for (int i = 0; i < 8; ++i) {
      const int m = i + 8 * hi;
      float iv = sigm(acc[0][i] + bi);
      float fv = sigm(acc[1][i] + bf_);
      float gv = tanh_f(acc[2][i] + bg);
      float ov = sigm(acc[3][i] + bo);
      float cn = fv * cst[i] + iv * gv;
      float hv = ov * tanh_f(cn);
      cst[i] = cn;
      unsigned short hb = f2bf(hv);
      abuf[m][KX + c] = hb;                                    // h for next step
      if (WRITE_Y) {
        gstore_b16(&y[((size_t)(rBase + m) * S_ + t) * 512 + dir * 256 + c], hb);
      } else if (it == S_ - 1) {
        gstore_f32(&hfin[(size_t)(rBase + m) * 512 + dir * 256 + c], hv);
      }
    }
    // next iteration's staging touches only the x region; the post-staging
    // barrier orders these h writes before the next WMMA reads.
  }
}

// -------------------------------- FC head ----------------------------------
__global__ void fc_kernel(const float* __restrict__ hfin,
                          const float* __restrict__ fc_w,
                          const float* __restrict__ fc_b,
                          float* __restrict__ out) {
  const int b = blockIdx.x;
  const int tid = threadIdx.x;   // 128 threads
  float s = 0.0f;
  for (int k = tid; k < 512; k += 128) s += hfin[(size_t)b * 512 + k] * fc_w[k];
  for (int o = 16; o > 0; o >>= 1) s += __shfl_down(s, o, 32);
  __shared__ float red[4];
  if ((tid & 31) == 0) red[tid >> 5] = s;
  __syncthreads();
  if (tid == 0) out[b] = red[0] + red[1] + red[2] + red[3] + fc_b[0];
}

// ------------------------------ host launcher ------------------------------
extern "C" void kernel_launch(void* const* d_in, const int* in_sizes, int n_in,
                              void* d_out, int out_size, void* d_ws, size_t ws_size,
                              hipStream_t stream) {
  (void)in_sizes; (void)n_in; (void)out_size; (void)ws_size;

  const int*   x        = (const int*)  d_in[0];
  const float* emb      = (const float*)d_in[1];
  const float* w_ih_l0  = (const float*)d_in[2];
  const float* w_hh_l0  = (const float*)d_in[3];
  const float* b_ih_l0  = (const float*)d_in[4];
  const float* b_hh_l0  = (const float*)d_in[5];
  const float* w_ih_l0r = (const float*)d_in[6];
  const float* w_hh_l0r = (const float*)d_in[7];
  const float* b_ih_l0r = (const float*)d_in[8];
  const float* b_hh_l0r = (const float*)d_in[9];
  const float* w_ih_l1  = (const float*)d_in[10];
  const float* w_hh_l1  = (const float*)d_in[11];
  const float* b_ih_l1  = (const float*)d_in[12];
  const float* b_hh_l1  = (const float*)d_in[13];
  const float* w_ih_l1r = (const float*)d_in[14];
  const float* w_hh_l1r = (const float*)d_in[15];
  const float* b_ih_l1r = (const float*)d_in[16];
  const float* b_hh_l1r = (const float*)d_in[17];
  const float* fc_w     = (const float*)d_in[18];
  const float* fc_b     = (const float*)d_in[19];

  // workspace layout (~101.3 MB total)
  char* ws = (char*)d_ws;
  const size_t SZ_XE  = (size_t)B_ * S_ * E_ * 2;          // 32 MB  bf16
  const size_t SZ_Y   = (size_t)B_ * S_ * 512 * 2;         // 64 MB  bf16
  const size_t SZ_W0  = (size_t)FOURH * (E_ + 256) * 2;    // 1  MB  bf16
  const size_t SZ_W1  = (size_t)FOURH * (512 + 256) * 2;   // 1.5 MB bf16
  const size_t OFF_XE   = 0;
  const size_t OFF_Y    = OFF_XE + SZ_XE;
  const size_t OFF_W0F  = OFF_Y + SZ_Y;
  const size_t OFF_W0B  = OFF_W0F + SZ_W0;
  const size_t OFF_W1F  = OFF_W0B + SZ_W0;
  const size_t OFF_W1B  = OFF_W1F + SZ_W1;
  const size_t OFF_BIAS = OFF_W1B + SZ_W1;                 // 4 x 1024 f32
  const size_t OFF_HFIN = OFF_BIAS + (size_t)4 * FOURH * 4;

  unsigned short* xe   = (unsigned short*)(ws + OFF_XE);
  unsigned short* ybuf = (unsigned short*)(ws + OFF_Y);
  unsigned short* W0f  = (unsigned short*)(ws + OFF_W0F);
  unsigned short* W0b  = (unsigned short*)(ws + OFF_W0B);
  unsigned short* W1f  = (unsigned short*)(ws + OFF_W1F);
  unsigned short* W1b  = (unsigned short*)(ws + OFF_W1B);
  float* bias0f = (float*)(ws + OFF_BIAS);
  float* bias0b = bias0f + FOURH;
  float* bias1f = bias0b + FOURH;
  float* bias1b = bias1f + FOURH;
  float* hfin   = (float*)(ws + OFF_HFIN);

  // 1) embedding -> bf16
  {
    long long total = (long long)B_ * S_ * (E_ / 4);
    embed_kernel<<<(unsigned)((total + 255) / 256), 256, 0, stream>>>(x, emb, xe);
  }
  // 2) pack weights / biases
  pack_w_kernel<<<512, 256, 0, stream>>>(w_ih_l0,  w_hh_l0,  W0f, 256);
  pack_w_kernel<<<512, 256, 0, stream>>>(w_ih_l0r, w_hh_l0r, W0b, 256);
  pack_w_kernel<<<768, 256, 0, stream>>>(w_ih_l1,  w_hh_l1,  W1f, 512);
  pack_w_kernel<<<768, 256, 0, stream>>>(w_ih_l1r, w_hh_l1r, W1b, 512);
  pack_b_kernel<<<4, 256, 0, stream>>>(b_ih_l0,  b_hh_l0,  bias0f);
  pack_b_kernel<<<4, 256, 0, stream>>>(b_ih_l0r, b_hh_l0r, bias0b);
  pack_b_kernel<<<4, 256, 0, stream>>>(b_ih_l1,  b_hh_l1,  bias1f);
  pack_b_kernel<<<4, 256, 0, stream>>>(b_ih_l1r, b_hh_l1r, bias1b);

  // 3) layer 0, both directions (writes y [B][S][512] bf16)
  lstm_dir_kernel<256, true><<<dim3(B_ / 16, 2), 512, 0, stream>>>(
      xe, W0f, W0b, bias0f, bias0b, ybuf, hfin);

  // 4) layer 1, both directions (writes only final h [B][512] f32)
  lstm_dir_kernel<512, false><<<dim3(B_ / 16, 2), 512, 0, stream>>>(
      ybuf, W1f, W1b, bias1f, bias1b, ybuf, hfin);

  // 5) FC head -> out[B] f32
  fc_kernel<<<B_, 128, 0, stream>>>(hfin, fc_w, fc_b, (float*)d_out);
}
